// PlainDETRTransformer_13030930776472
// MI455X (gfx1250) — matrix operand order repaired and checked by reading, not compiled, not run on hardware
//
#include <hip/hip_runtime.h>
#include <hip/hip_bf16.h>
#include <math.h>

// ---------------------------------------------------------------------------
// Problem constants (from the reference)
// ---------------------------------------------------------------------------
#define LB   6
#define DD   256
#define NHD  8
#define HDD  32
#define DFFD 1024
#define RPED 512
#define BB   2
#define NQD  300
#define FHW  64
#define NKEY 4096           // FH*FW
#define ROWS (BB * NQD)     // 600
#define SCALE_F 0.17677669529663687f  // 1/sqrt(32)

typedef __attribute__((ext_vector_type(16))) __bf16 v16bf;
typedef __attribute__((ext_vector_type(8)))  float  v8f;
typedef __attribute__((ext_vector_type(4)))  unsigned int v4u;
typedef __attribute__((ext_vector_type(8)))  int          v8i;
typedef __attribute__((ext_vector_type(4)))  int          v4i;

__device__ __forceinline__ unsigned short f2bf(float f) {
    unsigned int u = __float_as_uint(f);
    unsigned int r = u + 0x7FFFu + ((u >> 16) & 1u);   // round-to-nearest-even
    return (unsigned short)(r >> 16);
}
__device__ __forceinline__ unsigned int pk2bf(float a, float b) {
    return ((unsigned int)f2bf(b) << 16) | (unsigned int)f2bf(a);
}

// ---------------------------------------------------------------------------
// Generic GEMM:  C[M,N] = act(A[M,K] @ W[N,K]^T + bias[N])
// block = 128 threads (4 waves), each wave one 16x16 tile -> 64x16 per block.
// A tile: branchless vectorized global b128 loads -> packed bf16 LDS.
// W tile: fetched by the Tensor Data Mover (async DMA, TENSORcnt) as fp32,
//         then converted to bf16 in LDS. N is always a multiple of 16 here.
// ---------------------------------------------------------------------------
__global__ __launch_bounds__(128) void gemm_kernel(
    const float* __restrict__ A, const float* __restrict__ W,
    const float* __restrict__ bias, float* __restrict__ C,
    int M, int N, int K, int doRelu)
{
    __shared__ __align__(16) unsigned short As[64 * 32];
    __shared__ __align__(16) unsigned short Ws[16 * 32];
    __shared__ __align__(16) float          Wsf[16 * 32];

    int tid  = threadIdx.x;
    int wave = tid >> 5;
    int lane = tid & 31;
    int half = lane >> 4;
    int lc   = lane & 15;
    int m0 = blockIdx.x * 64;
    int n0 = blockIdx.y * 16;

    int arow = tid >> 1;            // A staging role: row 0..63
    int akk  = (tid & 1) * 16;      // 16 floats per thread

    v8f acc = {};

    for (int k0 = 0; k0 < K; k0 += 32) {
        // ---- W tile via Tensor Data Mover (wave 0 only; EXEC ignored) ----
        if (wave == 0) {
            unsigned long long ga =
                (unsigned long long)(const char*)(W + (size_t)n0 * K + k0);
            unsigned ldsa = (unsigned)(unsigned long long)(void*)Wsf;
            int dim0 = K - k0;                  // K extent left (OOB -> 0)
            v4u g0;
            g0[0] = 1u;                                         // count=1
            g0[1] = ldsa;                                       // lds_addr
            g0[2] = (unsigned)ga;                               // gaddr lo
            g0[3] = (unsigned)((ga >> 32) & 0x1FFFFFFu) | (2u << 30); // type=2
            v8i g1;
            g1[0] = 0x20000;                                    // data_size=4B
            g1[1] = (int)((unsigned)(dim0 & 0xFFFF) << 16);     // tensor_dim0 lo
            g1[2] = (int)(((unsigned)(dim0 >> 16) & 0xFFFFu) |
                          (16u << 16));                         // dim0 hi | tensor_dim1=16
            g1[3] = (int)(32u << 16);                           // tile_dim0 = 32
            g1[4] = 16;                                         // tile_dim1 = 16
            g1[5] = K;                                          // tensor_dim0_stride
            g1[6] = 0;
            g1[7] = 0;
            v4i z4 = {0, 0, 0, 0};
            v8i z8 = {0, 0, 0, 0, 0, 0, 0, 0};
            __builtin_amdgcn_tensor_load_to_lds(g0, g1, z4, z4, z8, 0);
        }

        // ---- A tile: 64x32 fp32 -> bf16, branchless (row-clamped) ----
        {
            int gr = m0 + arow;
            if (gr >= M) gr = M - 1;   // clamped rows feed only unstored outputs
            const float4* s = (const float4*)(A + (size_t)gr * K + k0 + akk);
            float4 f0 = s[0], f1 = s[1], f2 = s[2], f3 = s[3];
            uint4 lo, hi;
            lo.x = pk2bf(f0.x, f0.y); lo.y = pk2bf(f0.z, f0.w);
            lo.z = pk2bf(f1.x, f1.y); lo.w = pk2bf(f1.z, f1.w);
            hi.x = pk2bf(f2.x, f2.y); hi.y = pk2bf(f2.z, f2.w);
            hi.z = pk2bf(f3.x, f3.y); hi.w = pk2bf(f3.z, f3.w);
            uint4* d = (uint4*)(As + arow * 32 + akk);
            d[0] = lo;
            d[1] = hi;
        }

        if (wave == 0) __builtin_amdgcn_s_wait_tensorcnt(0);
        __syncthreads();               // As + Wsf ready

        // ---- convert W tile fp32 -> bf16 (LDS -> LDS), 4 elems/thread ----
        {
            int idx = tid * 4;
            float4 f = *(const float4*)(Wsf + idx);
            uint2 u;
            u.x = pk2bf(f.x, f.y);
            u.y = pk2bf(f.z, f.w);
            *(uint2*)(Ws + idx) = u;
        }
        __syncthreads();               // Ws ready

        // A fragment: row m = wave*16 + lc; K elems half*16 .. half*16+15
        v16bf af;
        {
            const unsigned short* p = As + (wave * 16 + lc) * 32;
            uint4* d = reinterpret_cast<uint4*>(&af);
            d[0] = *(const uint4*)(p + half * 16);
            d[1] = *(const uint4*)(p + half * 16 + 8);
        }
        // B fragment (B = W^T): col n = lc; K elems half*16 .. half*16+15
        v16bf bfrag;
        {
            const unsigned short* p = Ws + lc * 32 + half * 16;
            uint4* d = reinterpret_cast<uint4*>(&bfrag);
            d[0] = ((const uint4*)p)[0];
            d[1] = ((const uint4*)p)[1];
        }
        acc = __builtin_amdgcn_wmma_f32_16x16x32_bf16(
                  false, af, false, bfrag, (short)0, acc, false, false);
        __syncthreads();               // safe to overwrite tiles
    }

    int col = n0 + lc;                 // N multiple of 16 -> always valid
    float bv = bias ? bias[col] : 0.f;
#pragma unroll
    for (int r = 0; r < 8; ++r) {
        int row = m0 + wave * 16 + r + 8 * half;   // C layout: M = r + 8*half
        if (row < M) {
            float v = acc[r] + bv;
            if (doRelu) v = fmaxf(v, 0.f);
            C[(size_t)row * N + col] = v;
        }
    }
}

// ---------------------------------------------------------------------------
// Flash attention with optional decomposed RPE bias.
//   Qb  bf16 [B*NH][NQ][32]          (head-major)
//   Kb  bf16 [B*NH][NK][32]
//   Vt  bf16 [B*NH][32][NKS]         (key-major / transposed V)
//   O   fp32 [B][NQ][256]            (heads re-concatenated)
// One wave per (b,h,16-query tile); logits = scale*QK^T (+ rpe_x + rpe_y).
// ---------------------------------------------------------------------------
__global__ __launch_bounds__(32) void attn_kernel(
    const unsigned short* __restrict__ Qb,
    const unsigned short* __restrict__ Kb,
    const unsigned short* __restrict__ Vt,
    const float* __restrict__ rpex, const float* __restrict__ rpey,
    float* __restrict__ O,
    int NQ_, int NK_, int NKS_, float scale)
{
    int bh = blockIdx.x;
    int b  = bh / NHD;
    int h  = bh % NHD;
    int qbase = blockIdx.y * 16;
    int lane  = threadIdx.x;
    int half  = lane >> 4;
    int lc    = lane & 15;

    // Q fragment (A operand): row q = qbase + lc (clamped)
    v16bf qf;
    {
        int q = qbase + lc; if (q > NQ_ - 1) q = NQ_ - 1;
        const unsigned short* p = Qb + ((size_t)bh * NQ_ + q) * 32;
        uint4* d = reinterpret_cast<uint4*>(&qf);
        d[0] = *(const uint4*)(p + half * 16);
        d[1] = *(const uint4*)(p + half * 16 + 8);
    }

    v8f o0 = {}, o1 = {};
    float m8[8], l8[8];
#pragma unroll
    for (int r = 0; r < 8; ++r) { m8[r] = -1e30f; l8[r] = 0.f; }

    __shared__ __align__(16) unsigned short Ps[16 * 32];
    const v8f zeroc = {};

    for (int kb = 0; kb < NKS_; kb += 32) {
        float la[8], lb[8];
#pragma unroll
        for (int sub = 0; sub < 2; ++sub) {
            int kt = kb + sub * 16;
            v16bf kf;
            {
                int key = kt + lc; if (key > NK_ - 1) key = NK_ - 1;
                const unsigned short* p =
                    Kb + ((size_t)bh * NK_ + key) * 32 + half * 16;
                uint4* d = reinterpret_cast<uint4*>(&kf);
                d[0] = ((const uint4*)p)[0];
                d[1] = ((const uint4*)p)[1];
            }
            v8f s = __builtin_amdgcn_wmma_f32_16x16x32_bf16(
                        false, qf, false, kf, (short)0, zeroc, false, false);
            int key = kt + lc;                  // column of this lane
            float* dst = sub ? lb : la;
#pragma unroll
            for (int r = 0; r < 8; ++r) {
                int q = qbase + r + 8 * half;
                float v;
                if (key < NK_) {
                    v = s[r] * scale;
                    if (rpex) {
                        int qc = (q < NQ_) ? q : NQ_ - 1;
                        size_t rb = ((size_t)(b * NQD + qc)) * 64;
                        v += rpex[(rb + (key & 63)) * NHD + h];
                        v += rpey[(rb + (key >> 6)) * NHD + h];
                    }
                } else {
                    v = -1e30f;
                }
                dst[r] = v;
            }
        }
        // online softmax over this 32-key block
#pragma unroll
        for (int r = 0; r < 8; ++r) {
            float mx = fmaxf(la[r], lb[r]);
#pragma unroll
            for (int d = 1; d < 16; d <<= 1)
                mx = fmaxf(mx, __shfl_xor(mx, d, 32));
            float mnew  = fmaxf(m8[r], mx);
            float alpha = __expf(m8[r] - mnew);
            float pa = __expf(la[r] - mnew);
            float pb = __expf(lb[r] - mnew);
            float rs = pa + pb;
#pragma unroll
            for (int d = 1; d < 16; d <<= 1)
                rs += __shfl_xor(rs, d, 32);
            l8[r] = l8[r] * alpha + rs;
            m8[r] = mnew;
            o0[r] *= alpha;
            o1[r] *= alpha;
            int prow = r + 8 * half;
            Ps[prow * 32 + lc]      = f2bf(pa);
            Ps[prow * 32 + 16 + lc] = f2bf(pb);
        }
        // make cross-lane LDS P tile visible before reading it back
        asm volatile("s_wait_dscnt 0x0" ::: "memory");

        // P as A operand (16 q x 32 keys)
        v16bf pf;
        {
            const unsigned short* p = Ps + lc * 32;
            uint4* d = reinterpret_cast<uint4*>(&pf);
            d[0] = *(const uint4*)(p + half * 16);
            d[1] = *(const uint4*)(p + half * 16 + 8);
        }
        // V^T B operands for hd 0..15 and 16..31 (key-major, contiguous)
        v16bf vf0, vf1;
        {
            const unsigned short* p =
                Vt + ((size_t)bh * 32 + lc) * NKS_ + kb + half * 16;
            uint4* d = reinterpret_cast<uint4*>(&vf0);
            d[0] = ((const uint4*)p)[0];
            d[1] = ((const uint4*)p)[1];
            p += (size_t)16 * NKS_;   // hd = 16 + lc
            uint4* e = reinterpret_cast<uint4*>(&vf1);
            e[0] = ((const uint4*)p)[0];
            e[1] = ((const uint4*)p)[1];
        }
        o0 = __builtin_amdgcn_wmma_f32_16x16x32_bf16(
                 false, pf, false, vf0, (short)0, o0, false, false);
        o1 = __builtin_amdgcn_wmma_f32_16x16x32_bf16(
                 false, pf, false, vf1, (short)0, o1, false, false);
    }

#pragma unroll
    for (int r = 0; r < 8; ++r) {
        int q = qbase + r + 8 * half;
        if (q < NQ_) {
            float inv = 1.f / l8[r];
            size_t base = ((size_t)(b * NQD + q)) * DD + h * HDD;
            O[base + lc]      = o0[r] * inv;
            O[base + 16 + lc] = o1[r] * inv;
        }
    }
}

// ---------------------------------------------------------------------------
// Fused RPE MLP:  out[row, 0..7] = relu(d[row,0..1] @ W1^T + b1) @ W2^T
// one block per row, 128 threads, 512 hidden.
// ---------------------------------------------------------------------------
__global__ __launch_bounds__(128) void rpe_mlp_kernel(
    const float* __restrict__ d2, const float* __restrict__ w1,
    const float* __restrict__ b1, const float* __restrict__ w2,
    float* __restrict__ out)
{
    int row = blockIdx.x;
    int tid = threadIdx.x;
    float d0 = d2[(size_t)row * 2 + 0];
    float d1 = d2[(size_t)row * 2 + 1];
    float acc[NHD];
#pragma unroll
    for (int hh = 0; hh < NHD; ++hh) acc[hh] = 0.f;
#pragma unroll
    for (int i = 0; i < 4; ++i) {
        int j = tid + i * 128;
        float hv = fmaxf(w1[j * 2] * d0 + w1[j * 2 + 1] * d1 + b1[j], 0.f);
#pragma unroll
        for (int hh = 0; hh < NHD; ++hh)
            acc[hh] += w2[hh * RPED + j] * hv;
    }
    __shared__ float red[NHD * 128];
#pragma unroll
    for (int hh = 0; hh < NHD; ++hh) red[hh * 128 + tid] = acc[hh];
    __syncthreads();
    for (int s = 64; s > 0; s >>= 1) {
        if (tid < s) {
#pragma unroll
            for (int hh = 0; hh < NHD; ++hh)
                red[hh * 128 + tid] += red[hh * 128 + tid + s];
        }
        __syncthreads();
    }
    if (tid < NHD) out[(size_t)row * NHD + tid] = red[tid * 128];
}

// ---------------------------------------------------------------------------
// dx/dy builder: box corners minus pixel centers, per (b,q,pos)
// ---------------------------------------------------------------------------
__global__ void build_delta_kernel(const float* __restrict__ ref,
                                   float* __restrict__ dx,
                                   float* __restrict__ dy, int total)
{
    int i = blockIdx.x * blockDim.x + threadIdx.x;
    if (i >= total) return;
    int p  = i & 63;
    int qb = i >> 6;                      // b*NQ + q
    const float* r = ref + (size_t)qb * 4;
    float cx = r[0], cy = r[1], w = r[2], h = r[3];
    float pc = (p + 0.5f) * 16.f;
    dx[(size_t)i * 2 + 0] = (cx - 0.5f * w) - pc;
    dx[(size_t)i * 2 + 1] = (cx + 0.5f * w) - pc;
    dy[(size_t)i * 2 + 0] = (cy - 0.5f * h) - pc;
    dy[(size_t)i * 2 + 1] = (cy + 0.5f * h) - pc;
}

// ---------------------------------------------------------------------------
// LayerNorm with residual, in place: x = LN(x + t) * w + b   (D=256)
// ---------------------------------------------------------------------------
__global__ __launch_bounds__(256) void ln_res_kernel(
    float* __restrict__ x, const float* __restrict__ t,
    const float* __restrict__ w, const float* __restrict__ b)
{
    int row = blockIdx.x, tid = threadIdx.x;
    __shared__ float red[256];
    size_t idx = (size_t)row * DD + tid;
    float v = x[idx] + t[idx];
    red[tid] = v; __syncthreads();
    for (int s = 128; s > 0; s >>= 1) {
        if (tid < s) red[tid] += red[tid + s];
        __syncthreads();
    }
    float mean = red[0] * (1.f / DD);
    __syncthreads();
    float d = v - mean;
    red[tid] = d * d; __syncthreads();
    for (int s = 128; s > 0; s >>= 1) {
        if (tid < s) red[tid] += red[tid + s];
        __syncthreads();
    }
    float var = red[0] * (1.f / DD);
    float rst = rsqrtf(var + 1e-5f);
    x[idx] = d * rst * w[tid] + b[tid];
}

// ---------------------------------------------------------------------------
// Elementwise add:  c = a + b
// ---------------------------------------------------------------------------
__global__ void add2_kernel(float* __restrict__ c, const float* __restrict__ a,
                            const float* __restrict__ b, int n)
{
    int i = blockIdx.x * blockDim.x + threadIdx.x;
    if (i < n) c[i] = a[i] + b[i];
}

// ---------------------------------------------------------------------------
// Pack fp32 rows into head-major bf16:
//   dst[((b*NH+h)*NL + q)*32 + hd] = bf16(src[(b*NL+q)*stride + colOff + h*32 + hd])
// ---------------------------------------------------------------------------
__global__ void pack_rows_kernel(const float* __restrict__ src,
                                 unsigned short* __restrict__ dst,
                                 int NL, int stride, int colOff, int total)
{
    int i = blockIdx.x * blockDim.x + threadIdx.x;
    if (i >= total) return;
    int hd = i & 31;
    int t  = i >> 5;
    int q  = t % NL; t /= NL;
    int h  = t % NHD;
    int b  = t / NHD;
    dst[i] = f2bf(src[((size_t)(b * NL + q)) * stride + colOff + h * HDD + hd]);
}

// ---------------------------------------------------------------------------
// Pack V into key-major transposed bf16:
//   dst[((b*NH+h)*32 + hd)*NKS + n] = bf16(src[(b*NK+n)*stride + h*32 + hd])
// (zero-pad n >= NK)
// ---------------------------------------------------------------------------
__global__ void pack_vt_kernel(const float* __restrict__ src,
                               unsigned short* __restrict__ dst,
                               int NK, int NKS, int stride, int total)
{
    int i = blockIdx.x * blockDim.x + threadIdx.x;
    if (i >= total) return;
    int n = i % NKS;
    int t = i / NKS;
    int hd = t & 31; t >>= 5;
    int h  = t % NHD;
    int b  = t / NHD;
    dst[i] = (n < NK)
        ? f2bf(src[((size_t)(b * NK + n)) * stride + h * HDD + hd])
        : (unsigned short)0;
}

// ---------------------------------------------------------------------------
// Host side
// ---------------------------------------------------------------------------
extern "C" void kernel_launch(void* const* d_in, const int* in_sizes, int n_in,
                              void* d_out, int out_size, void* d_ws, size_t ws_size,
                              hipStream_t stream) {
    (void)in_sizes; (void)n_in; (void)out_size; (void)ws_size;

    const float* tgt        = (const float*)d_in[0];
    const float* query_pos  = (const float*)d_in[1];
    const float* refpts     = (const float*)d_in[2];
    const float* src        = (const float*)d_in[3];
    const float* src_pos    = (const float*)d_in[4];
    const float* sa_in_w    = (const float*)d_in[6];
    const float* sa_in_b    = (const float*)d_in[7];
    const float* sa_out_w   = (const float*)d_in[8];
    const float* sa_out_b   = (const float*)d_in[9];
    const float* n1_w       = (const float*)d_in[10];
    const float* n1_b       = (const float*)d_in[11];
    const float* ca_q_w     = (const float*)d_in[12];
    const float* ca_q_b     = (const float*)d_in[13];
    const float* ca_k_w     = (const float*)d_in[14];
    const float* ca_k_b     = (const float*)d_in[15];
    const float* ca_v_w     = (const float*)d_in[16];
    const float* ca_v_b     = (const float*)d_in[17];
    const float* ca_p_w     = (const float*)d_in[18];
    const float* ca_p_b     = (const float*)d_in[19];
    const float* cpb1_w1    = (const float*)d_in[20];
    const float* cpb1_b1    = (const float*)d_in[21];
    const float* cpb1_w2    = (const float*)d_in[22];
    const float* cpb2_w1    = (const float*)d_in[23];
    const float* cpb2_b1    = (const float*)d_in[24];
    const float* cpb2_w2    = (const float*)d_in[25];
    const float* n2_w       = (const float*)d_in[26];
    const float* n2_b       = (const float*)d_in[27];
    const float* ffn_w1     = (const float*)d_in[28];
    const float* ffn_b1     = (const float*)d_in[29];
    const float* ffn_w2     = (const float*)d_in[30];
    const float* ffn_b2     = (const float*)d_in[31];
    const float* n3_w       = (const float*)d_in[32];
    const float* n3_b       = (const float*)d_in[33];

    float* out = (float*)d_out;   // decoder state lives in d_out

    // ---- workspace layout ----
    char* wsp = (char*)d_ws;
    auto alloc = [&](size_t bytes) {
        char* p = wsp;
        wsp += (bytes + 255) & ~(size_t)255;
        return p;
    };
    float*          kv   = (float*)alloc((size_t)BB * NKEY * DD * 4);
    float*          qin  = (float*)alloc((size_t)ROWS * DD * 4);
    float*          bufQ = (float*)alloc((size_t)ROWS * DD * 4);
    float*          bufA = (float*)alloc((size_t)BB * NKEY * DD * 4);  // self QK (600x512) / cross K
    float*          bufB = (float*)alloc((size_t)BB * NKEY * DD * 4);  // self V / cross V
    unsigned short* Qb   = (unsigned short*)alloc((size_t)BB * NHD * NQD * HDD * 2);
    unsigned short* Kb   = (unsigned short*)alloc((size_t)BB * NHD * NKEY * HDD * 2);
    unsigned short* Vtb  = (unsigned short*)alloc((size_t)BB * NHD * HDD * NKEY * 2);
    float*          Obuf = (float*)alloc((size_t)ROWS * DD * 4);
    float*          Tbuf = (float*)alloc((size_t)ROWS * DD * 4);
    float*          Hbuf = (float*)alloc((size_t)ROWS * DFFD * 4);
    float*          dx   = (float*)alloc((size_t)ROWS * FHW * 2 * 4);
    float*          dy   = (float*)alloc((size_t)ROWS * FHW * 2 * 4);
    float*          rpex = (float*)alloc((size_t)ROWS * FHW * NHD * 4);
    float*          rpey = (float*)alloc((size_t)ROWS * FHW * NHD * 4);

    auto cdiv = [](int a, int b) { return (a + b - 1) / b; };
    auto gemm = [&](const float* A, const float* W, const float* bias,
                    float* C, int M, int Nn, int K, int relu) {
        dim3 g(cdiv(M, 64), cdiv(Nn, 16));
        gemm_kernel<<<g, 128, 0, stream>>>(A, W, bias, C, M, Nn, K, relu);
    };

    // state = tgt
    (void)hipMemcpyAsync(out, tgt, (size_t)ROWS * DD * 4,
                         hipMemcpyDeviceToDevice, stream);
    // kv = src + src_pos_embed
    {
        int n = BB * NKEY * DD;
        add2_kernel<<<cdiv(n, 256), 256, 0, stream>>>(kv, src, src_pos, n);
    }
    // dx/dy (layer-invariant)
    {
        int total = ROWS * FHW;
        build_delta_kernel<<<cdiv(total, 256), 256, 0, stream>>>(refpts, dx, dy, total);
    }

    const int qTiles = cdiv(NQD, 16);   // 19
    for (int l = 0; l < LB; ++l) {
        const float* saw = sa_in_w  + (size_t)l * 3 * DD * DD;
        const float* sab = sa_in_b  + (size_t)l * 3 * DD;
        const float* sow = sa_out_w + (size_t)l * DD * DD;
        const float* sob = sa_out_b + (size_t)l * DD;

        // ---------------- self attention ----------------
        {
            int n = ROWS * DD;
            add2_kernel<<<cdiv(n, 256), 256, 0, stream>>>(qin, out, query_pos, n);
        }
        gemm(qin, saw, sab, bufA, ROWS, 2 * DD, DD, 0);                    // [q|k]
        gemm(out, saw + (size_t)2 * DD * DD, sab + 2 * DD, bufB,
             ROWS, DD, DD, 0);                                             // v
        {
            int tq = BB * NHD * NQD * HDD;
            pack_rows_kernel<<<cdiv(tq, 256), 256, 0, stream>>>(
                bufA, Qb, NQD, 2 * DD, 0, tq);
            pack_rows_kernel<<<cdiv(tq, 256), 256, 0, stream>>>(
                bufA, Kb, NQD, 2 * DD, DD, tq);
            int NKS = 320;                                                 // pad 300->320
            int tv = BB * NHD * HDD * NKS;
            pack_vt_kernel<<<cdiv(tv, 256), 256, 0, stream>>>(
                bufB, Vtb, NQD, NKS, DD, tv);
            attn_kernel<<<dim3(BB * NHD, qTiles), 32, 0, stream>>>(
                Qb, Kb, Vtb, nullptr, nullptr, Obuf, NQD, NQD, NKS, SCALE_F);
        }
        gemm(Obuf, sow, sob, Tbuf, ROWS, DD, DD, 0);
        ln_res_kernel<<<ROWS, 256, 0, stream>>>(
            out, Tbuf, n1_w + (size_t)l * DD, n1_b + (size_t)l * DD);

        // ---------------- cross attention ----------------
        {
            int n = ROWS * DD;
            add2_kernel<<<cdiv(n, 256), 256, 0, stream>>>(qin, out, query_pos, n);
        }
        gemm(qin, ca_q_w + (size_t)l * DD * DD, ca_q_b + (size_t)l * DD,
             bufQ, ROWS, DD, DD, 0);
        gemm(kv,  ca_k_w + (size_t)l * DD * DD, ca_k_b + (size_t)l * DD,
             bufA, BB * NKEY, DD, DD, 0);
        gemm(src, ca_v_w + (size_t)l * DD * DD, ca_v_b + (size_t)l * DD,
             bufB, BB * NKEY, DD, DD, 0);
        {
            int tq = BB * NHD * NQD * HDD;
            pack_rows_kernel<<<cdiv(tq, 256), 256, 0, stream>>>(
                bufQ, Qb, NQD, DD, 0, tq);
            int tk = BB * NHD * NKEY * HDD;
            pack_rows_kernel<<<cdiv(tk, 256), 256, 0, stream>>>(
                bufA, Kb, NKEY, DD, 0, tk);
            int tv = BB * NHD * HDD * NKEY;
            pack_vt_kernel<<<cdiv(tv, 256), 256, 0, stream>>>(
                bufB, Vtb, NKEY, NKEY, DD, tv);
        }
        rpe_mlp_kernel<<<ROWS * FHW, 128, 0, stream>>>(
            dx, cpb1_w1 + (size_t)l * RPED * 2, cpb1_b1 + (size_t)l * RPED,
            cpb1_w2 + (size_t)l * NHD * RPED, rpex);
        rpe_mlp_kernel<<<ROWS * FHW, 128, 0, stream>>>(
            dy, cpb2_w1 + (size_t)l * RPED * 2, cpb2_b1 + (size_t)l * RPED,
            cpb2_w2 + (size_t)l * NHD * RPED, rpey);
        attn_kernel<<<dim3(BB * NHD, qTiles), 32, 0, stream>>>(
            Qb, Kb, Vtb, rpex, rpey, Obuf, NQD, NKEY, NKEY, SCALE_F);
        gemm(Obuf, ca_p_w + (size_t)l * DD * DD, ca_p_b + (size_t)l * DD,
             Tbuf, ROWS, DD, DD, 0);
        ln_res_kernel<<<ROWS, 256, 0, stream>>>(
            out, Tbuf, n2_w + (size_t)l * DD, n2_b + (size_t)l * DD);

        // ---------------- FFN ----------------
        gemm(out, ffn_w1 + (size_t)l * DFFD * DD, ffn_b1 + (size_t)l * DFFD,
             Hbuf, ROWS, DFFD, DD, 1);
        gemm(Hbuf, ffn_w2 + (size_t)l * DD * DFFD, ffn_b2 + (size_t)l * DD,
             Tbuf, ROWS, DD, DFFD, 0);
        ln_res_kernel<<<ROWS, 256, 0, stream>>>(
            out, Tbuf, n3_w + (size_t)l * DD, n3_b + (size_t)l * DD);
    }
}
